// BahdanauAttention_22376779612722
// MI455X (gfx1250) — matrix operand order repaired and checked by reading, not compiled, not run on hardware
//
#include <hip/hip_runtime.h>
#include <hip/hip_bf16.h>
#include <math.h>

// Problem constants (match reference)
#define BB 32
#define SS 2048
#define HH 1024

typedef __attribute__((ext_vector_type(16))) __bf16 v16bf;
typedef __attribute__((ext_vector_type(8)))  float  v8f;

// ---------------------------------------------------------------------------
// Kernel 1: repack Wh (HxH f32, row-major [k][n]) into bf16 WMMA B-fragments.
// Tile index = nt*32 + kt  (nt: 0..63 over N, kt: 0..31 over K).
// B-fragment layout (32x16, 16-bit): lane = n + 16*(kk>=16), elem e = kk&15.
// Each lane's 16 bf16 stored contiguously -> main kernel loads 32B/lane.
// ---------------------------------------------------------------------------
__global__ __launch_bounds__(32) void pack_wh(const float* __restrict__ Wh,
                                              __hip_bfloat16* __restrict__ whp) {
  int tile = blockIdx.x;               // 2048 tiles
  int nt = tile >> 5, kt = tile & 31;
  int lane = threadIdx.x;              // 0..31
  int n = nt * 16 + (lane & 15);
  int kbase = kt * 32 + ((lane >> 4) << 4);   // +16 for hi half-lanes
  __hip_bfloat16* dst = whp + (size_t)tile * 512 + (size_t)lane * 16;
#pragma unroll
  for (int e = 0; e < 16; ++e) {
    int k = kbase + e;
    dst[e] = __float2bfloat16(Wh[(size_t)k * HH + n]);
  }
}

// ---------------------------------------------------------------------------
// Kernel 2: dec_p[b][h] = decoder_state[b] @ Ws + bs  (f32, tiny)
//           cov_p[b]    = coverage[b] @ Wc + bc       (scalar per batch)
// ---------------------------------------------------------------------------
__global__ __launch_bounds__(256) void dec_cov(const float* __restrict__ dec,
                                               const float* __restrict__ Wsm,
                                               const float* __restrict__ bs,
                                               const float* __restrict__ cover,
                                               const float* __restrict__ Wc,
                                               const float* __restrict__ bc,
                                               float* __restrict__ decp,
                                               float* __restrict__ covp) {
  int b = blockIdx.x;
  int t = threadIdx.x;
  __shared__ float sd[HH];
  __shared__ float red[256];

  for (int k = t; k < HH; k += 256) sd[k] = dec[(size_t)b * HH + k];
  __syncthreads();

  for (int h = t; h < HH; h += 256) {
    float a = bs[h];
    for (int k = 0; k < HH; ++k) a = fmaf(sd[k], Wsm[(size_t)k * HH + h], a);
    decp[(size_t)b * HH + h] = a;
  }

  float c = 0.f;
  for (int s = t; s < SS; s += 256) c = fmaf(cover[(size_t)b * SS + s], Wc[s], c);
  red[t] = c;
  __syncthreads();
  for (int o = 128; o > 0; o >>= 1) {
    if (t < o) red[t] += red[t + o];
    __syncthreads();
  }
  if (t == 0) covp[b] = red[0] + bc[0];
}

// ---------------------------------------------------------------------------
// Kernel 3 (main, WMMA): per block = (batch b, 32 rows of S = 2 M-subtiles).
// Stage A (32x1024) as bf16 fragments in 64KB LDS; wave w owns N-columns
// [128w, 128w+128). Each B fragment load feeds TWO wmmas (one per M-subtile),
// halving Wh/L2 traffic vs M=16 and doubling compute per loaded byte.
// A-fragment layout (16x32, 16-bit): lane = m + 16*(kk>=16),
//   elem e = kk&15.
// C layout: acc element r <-> (M = r + 8*(lane>>4), N = lane&15).
// ---------------------------------------------------------------------------
__global__ __launch_bounds__(256) void attn_gemm(const float* __restrict__ enc,
                                                 const __hip_bfloat16* __restrict__ whp,
                                                 const float* __restrict__ bh,
                                                 const float* __restrict__ Vv,
                                                 const float* __restrict__ decp,
                                                 const float* __restrict__ covp,
                                                 float* __restrict__ scores) {
  // A fragments: [msub(2)][kt(32)][lane(32)][elem(16)] bf16 = 64 KB
  __shared__ __align__(32) __hip_bfloat16 sA[2 * 32 * 512];
  __shared__ float sScore[32];

  int b  = blockIdx.x >> 6;         // 64 s-tiles (of 32 rows) per batch
  int st = blockIdx.x & 63;
  int s0 = st * 32;
  int t = threadIdx.x;
  int wave = t >> 5;
  int lane = t & 31;

  if (t < 32) sScore[t] = 0.f;

  // Stage + convert A: 2048 slots; slot = (msub*32 + kt)*32 + l.
  for (int slot = t; slot < 2048; slot += 256) {
    int l = slot & 31;
    int ktm = slot >> 5;            // 0..63
    int kt = ktm & 31, msub = ktm >> 5;
    int m = l & 15, hi = l >> 4;
    const float* src = enc + ((size_t)b * SS + s0 + msub * 16 + m) * HH + kt * 32;
    __hip_bfloat16* dst = sA + (size_t)slot * 16;
#pragma unroll
    for (int e = 0; e < 8; ++e) dst[e] = __float2bfloat16(src[hi * 16 + e]);
#pragma unroll
    for (int e = 0; e < 8; ++e) dst[8 + e] = __float2bfloat16(src[hi * 16 + 8 + e]);
  }
  __syncthreads();

  v8f acc0[8], acc1[8];
#pragma unroll
  for (int i = 0; i < 8; ++i) {
    acc0[i] = (v8f){0.f, 0.f, 0.f, 0.f, 0.f, 0.f, 0.f, 0.f};
    acc1[i] = (v8f){0.f, 0.f, 0.f, 0.f, 0.f, 0.f, 0.f, 0.f};
  }

  const int n0w = wave * 8;  // first N-tile for this wave
  const v16bf* wp = (const v16bf*)(const void*)whp;  // 32 v16bf per tile

  for (int kt = 0; kt < 32; ++kt) {
    v16bf a0 = *(const v16bf*)(const void*)(sA + ((size_t)(kt * 32 + lane)) * 16);
    v16bf a1 = *(const v16bf*)(const void*)(sA + ((size_t)(1024 + kt * 32 + lane)) * 16);
#pragma unroll
    for (int ti = 0; ti < 8; ++ti) {
      int tile = (n0w + ti) * 32 + kt;
      v16bf bf = wp[(size_t)tile * 32 + lane];
      acc0[ti] = __builtin_amdgcn_wmma_f32_16x16x32_bf16(
          false, a0, false, bf, (short)0, acc0[ti], false, false);
      acc1[ti] = __builtin_amdgcn_wmma_f32_16x16x32_bf16(
          false, a1, false, bf, (short)0, acc1[ti], false, false);
    }
  }

  // Fused epilogue: tanh(enc_p + bh + dec_p + cov_p) * V, reduce over H.
  float cov = covp[b];
  float p0[8], p1[8];
#pragma unroll
  for (int r = 0; r < 8; ++r) { p0[r] = 0.f; p1[r] = 0.f; }
#pragma unroll
  for (int ti = 0; ti < 8; ++ti) {
    int h = (n0w + ti) * 16 + (lane & 15);
    float add = bh[h] + decp[(size_t)b * HH + h] + cov;
    float vh = Vv[h];
#pragma unroll
    for (int r = 0; r < 8; ++r) {
      p0[r] += tanhf(acc0[ti][r] + add) * vh;
      p1[r] += tanhf(acc1[ti][r] + add) * vh;
    }
  }
  int mbase = (lane >> 4) * 8;
#pragma unroll
  for (int r = 0; r < 8; ++r) {
    atomicAdd(&sScore[mbase + r], p0[r]);
    atomicAdd(&sScore[16 + mbase + r], p1[r]);
  }
  __syncthreads();

  if (t < 32) scores[(size_t)b * SS + s0 + t] = sScore[t];
}

// ---------------------------------------------------------------------------
// Kernel 4: per-batch softmax stats (max, sum of exp). bv dropped: softmax
// is invariant to a per-row constant and raw scores are never output.
// ---------------------------------------------------------------------------
__global__ __launch_bounds__(256) void softmax_stats(const float* __restrict__ scores,
                                                     float* __restrict__ stats) {
  int b = blockIdx.x, t = threadIdx.x;
  __shared__ float red[256];
  float m = -1e30f;
  for (int s = t; s < SS; s += 256) m = fmaxf(m, scores[(size_t)b * SS + s]);
  red[t] = m;
  __syncthreads();
  for (int o = 128; o > 0; o >>= 1) {
    if (t < o) red[t] = fmaxf(red[t], red[t + o]);
    __syncthreads();
  }
  m = red[0];
  __syncthreads();
  float sum = 0.f;
  for (int s = t; s < SS; s += 256) sum += __expf(scores[(size_t)b * SS + s] - m);
  red[t] = sum;
  __syncthreads();
  for (int o = 128; o > 0; o >>= 1) {
    if (t < o) red[t] += red[t + o];
    __syncthreads();
  }
  if (t == 0) { stats[2 * b] = m; stats[2 * b + 1] = red[0]; }
}

// ---------------------------------------------------------------------------
// Kernel 5: attention weights + new coverage.
// ---------------------------------------------------------------------------
__global__ __launch_bounds__(256) void attn_cov(const float* __restrict__ scores,
                                                const float* __restrict__ stats,
                                                const float* __restrict__ cover,
                                                float* __restrict__ out_attn,
                                                float* __restrict__ out_cov) {
  int b = blockIdx.x, t = threadIdx.x;
  float m = stats[2 * b];
  float inv = 1.f / stats[2 * b + 1];
  for (int s = t; s < SS; s += 256) {
    float w = __expf(scores[(size_t)b * SS + s] - m) * inv;
    out_attn[(size_t)b * SS + s] = w;
    out_cov[(size_t)b * SS + s] = cover[(size_t)b * SS + s] + w;
  }
}

// ---------------------------------------------------------------------------
// Kernel 6: context = sum_s attn[b,s] * enc[b,s,:]  (coalesced over h).
// ---------------------------------------------------------------------------
__global__ __launch_bounds__(128) void context_kernel(const float* __restrict__ enc,
                                                      const float* __restrict__ attn,
                                                      float* __restrict__ out_ctx) {
  int b = blockIdx.y;
  int h = blockIdx.x * 128 + threadIdx.x;
  __shared__ float wsh[128];
  float acc = 0.f;
  for (int s0 = 0; s0 < SS; s0 += 128) {
    __syncthreads();
    wsh[threadIdx.x] = attn[(size_t)b * SS + s0 + threadIdx.x];
    __syncthreads();
#pragma unroll 4
    for (int i = 0; i < 128; ++i)
      acc = fmaf(wsh[i], enc[((size_t)b * SS + s0 + i) * HH + h], acc);
  }
  out_ctx[(size_t)b * HH + h] = acc;
}

// ---------------------------------------------------------------------------
extern "C" void kernel_launch(void* const* d_in, const int* in_sizes, int n_in,
                              void* d_out, int out_size, void* d_ws, size_t ws_size,
                              hipStream_t stream) {
  const float* dec   = (const float*)d_in[0];   // (B,H)
  const float* enc   = (const float*)d_in[1];   // (B,S,H)
  const float* cover = (const float*)d_in[2];   // (B,S)
  const float* Wh    = (const float*)d_in[3];   // (H,H)
  const float* bh    = (const float*)d_in[4];   // (H,)
  const float* Wsm   = (const float*)d_in[5];   // (H,H)
  const float* bs    = (const float*)d_in[6];   // (H,)
  const float* Vv    = (const float*)d_in[7];   // (H,1)
  // d_in[8] = bv — softmax-invariant, not needed
  const float* Wc    = (const float*)d_in[9];   // (S,1)
  const float* bc    = (const float*)d_in[10];  // (1,)

  float* out_ctx  = (float*)d_out;              // B*H
  float* out_attn = out_ctx + BB * HH;          // B*S
  float* out_cov  = out_attn + BB * SS;         // B*S

  // Workspace layout (floats), keeps whp 32B-aligned:
  float* w      = (float*)d_ws;
  float* decp   = w;                      // B*H   = 32768
  float* covp   = decp + BB * HH;         // 64 (padded)
  float* stats  = covp + 64;              // 64 (2 per batch, padded)
  float* scores = stats + 64;             // B*S   = 65536
  __hip_bfloat16* whp = (__hip_bfloat16*)(scores + BB * SS);  // H*H bf16 = 2MB

  pack_wh<<<dim3(2048), dim3(32), 0, stream>>>(Wh, whp);
  dec_cov<<<dim3(BB), dim3(256), 0, stream>>>(dec, Wsm, bs, cover, Wc, bc, decp, covp);
  attn_gemm<<<dim3(BB * (SS / 32)), dim3(256), 0, stream>>>(
      enc, (const __hip_bfloat16*)whp, bh, Vv, decp, covp, scores);
  softmax_stats<<<dim3(BB), dim3(256), 0, stream>>>(scores, stats);
  attn_cov<<<dim3(BB), dim3(256), 0, stream>>>(scores, stats, cover, out_attn, out_cov);
  context_kernel<<<dim3(HH / 128, BB), dim3(128), 0, stream>>>(enc, out_attn, out_ctx);
}
